// Model_66013647340157
// MI455X (gfx1250) — matrix operand order repaired and checked by reading, not compiled, not run on hardware
//
#include <hip/hip_runtime.h>
#include <hip/hip_bf16.h>
#include <math.h>

// ---------------------------------------------------------------------------
// Problem constants: V=30000, E=300, H=512, W=20, L=512
// ---------------------------------------------------------------------------
#define LEN   512
#define EDIM  300
#define KP    320           // E padded to multiple of 32 for WMMA K-steps
#define HDIM  512
#define WROWS 20
#define GDIM  2048          // 4*H gate rows
#define EPSF  1e-8f

typedef __attribute__((ext_vector_type(16))) _Float16 v16h;
typedef __attribute__((ext_vector_type(8)))  _Float16 v8h;
typedef __attribute__((ext_vector_type(8)))  float    v8f;

// ---------------------------------------------------------------------------
// WMMA helpers: v_wmma_f32_16x16x32_f16 (wave32).
// A frag (16x32): lane holds row M=lane&15; elems 0..7 = K kb+(lane&16?8:0)+u,
//                 elems 8..15 = +16. Two aligned 16B loads.
// B frag (32x16): lane holds col N=lane&15; same K layout as A frag.
//                 Two aligned 16B loads.
// C/D: elem v of lane l -> (M = v + (l&16?8:0), N = l&15).
// ---------------------------------------------------------------------------
__device__ __forceinline__ v8f wmma_step(v16h a, v16h b, v8f c) {
  return __builtin_amdgcn_wmma_f32_16x16x32_f16(
      false, a, false, b, (short)0, c, false, false);
}

__device__ __forceinline__ v16h frag_a_h(const _Float16* __restrict__ row, int kb, int lane) {
  const int sel = (lane & 16) ? 8 : 0;
  v8h lo = *(const v8h*)(row + kb + sel);
  v8h hi = *(const v8h*)(row + kb + 16 + sel);
  return __builtin_shufflevector(lo, hi, 0, 1, 2, 3, 4, 5, 6, 7,
                                 8, 9, 10, 11, 12, 13, 14, 15);
}

__device__ __forceinline__ v16h frag_b_h(const _Float16* __restrict__ row, int kb, int lane) {
  const int sel = (lane & 16) ? 8 : 0;
  v8h lo = *(const v8h*)(row + kb + sel);
  v8h hi = *(const v8h*)(row + kb + 16 + sel);
  return __builtin_shufflevector(lo, hi, 0, 1, 2, 3, 4, 5, 6, 7,
                                 8, 9, 10, 11, 12, 13, 14, 15);
}

// A fragment from f32 source, columns scaled by w[h]^2 (for pair-max).
__device__ __forceinline__ v16h frag_a_w2(const float* __restrict__ row,
                                          const float* __restrict__ w, int kb, int lane) {
  const int sel = (lane & 16) ? 8 : 0;
  v16h f;
#pragma unroll
  for (int u = 0; u < 8; ++u) {
    int k0 = kb + sel + u, k1 = kb + 16 + sel + u;
    float w0 = w[k0], w1 = w[k1];
    f[u]     = (_Float16)(row[k0] * (w0 * w0));
    f[u + 8] = (_Float16)(row[k1] * (w1 * w1));
  }
  return f;
}

// ---------------------------------------------------------------------------
// 1) Embedding gather -> padded f16 X [512 x 320] (zero pad cols 300..319)
// ---------------------------------------------------------------------------
__global__ void gather_pack_kernel(const float* __restrict__ emb,
                                   const int* __restrict__ ids1,
                                   const int* __restrict__ ids2,
                                   _Float16* __restrict__ xh1,
                                   _Float16* __restrict__ xh2) {
  int idx = blockIdx.x * blockDim.x + threadIdx.x;
  const int total = 2 * LEN * KP;
  if (idx >= total) return;
  int which = idx / (LEN * KP);
  int rem   = idx - which * (LEN * KP);
  int r = rem / KP, c = rem - r * KP;
  const int* ids = which ? ids2 : ids1;
  _Float16* xh = which ? xh2 : xh1;
  float v = (c < EDIM) ? emb[(long)ids[r] * EDIM + c] : 0.f;
  xh[r * KP + c] = (_Float16)v;
}

// 2) Pack the 4 Wih [2048 x 300] f32 -> [2048 x 320] f16 (zero padded)
struct PackWArgs { const float* src[4]; _Float16* dst[4]; };

__global__ void pack_wih_kernel(PackWArgs args) {
  int idx = blockIdx.x * blockDim.x + threadIdx.x;
  const int per = GDIM * KP;
  if (idx >= 4 * per) return;
  int d = idx / per;
  int rem = idx - d * per;
  int r = rem / KP, c = rem - r * KP;
  float v = (c < EDIM) ? args.src[d][(long)r * EDIM + c] : 0.f;
  args.dst[d][r * KP + c] = (_Float16)v;
}

// 3) Pack q1p/q1n/q2p/q2n [512 x 512] f32 -> f16
struct PackQArgs { const float* src[4]; _Float16* dst[4]; };

__global__ void pack_q_kernel(PackQArgs args) {
  int idx = blockIdx.x * blockDim.x + threadIdx.x;
  const int per = LEN * HDIM;
  if (idx >= 4 * per) return;
  int d = idx / per;
  int rem = idx - d * per;
  args.dst[d][rem] = (_Float16)args.src[d][rem];
}

// ---------------------------------------------------------------------------
// 4) Input projections P_d[t, r] = sum_e X[t,e] * Wih_d[r,e]  (WMMA, f16 in)
//    M=512(T), N=2048(gate rows), K=320. One wave per 16x16 tile.
// ---------------------------------------------------------------------------
struct ProjArgs { const _Float16* X[4]; const _Float16* Wt[4]; float* P[4]; };

__global__ __launch_bounds__(32) void proj_gemm_kernel(ProjArgs args) {
  const int lane = threadIdx.x;
  const int nt = blockIdx.x;            // gate-row tile
  const int mt = blockIdx.y;            // time tile
  const int d  = blockIdx.z;            // direction
  const _Float16* arow = args.X[d]  + (mt * 16 + (lane & 15)) * KP;
  const _Float16* brow = args.Wt[d] + (nt * 16 + (lane & 15)) * KP;
  v8f acc = {};
#pragma unroll
  for (int kb = 0; kb < KP; kb += 32)
    acc = wmma_step(frag_a_h(arow, kb, lane), frag_b_h(brow, kb, lane), acc);
  float* P = args.P[d];
  const int M0 = (lane & 16) ? 8 : 0;
  const int n  = nt * 16 + (lane & 15);
#pragma unroll
  for (int v = 0; v < 8; ++v)
    P[(mt * 16 + M0 + v) * GDIM + n] = acc[v];
}

// ---------------------------------------------------------------------------
// 5) Sequential LSTM scan: one persistent 1024-thread workgroup / direction.
//    h in LDS; float4 (b128) reads of Whh rows and h.
// ---------------------------------------------------------------------------
struct LstmArgs { const float* P[4]; const float* Whh[4]; float* Hout[4]; int back[4]; };

__device__ __forceinline__ float sigm(float x) { return 1.0f / (1.0f + __expf(-x)); }

__global__ __launch_bounds__(1024) void lstm_kernel(LstmArgs args) {
  __shared__ __align__(16) float hbuf[HDIM];
  __shared__ float gI[HDIM], gF[HDIM], gG[HDIM], gO[HDIM];
  const int d = blockIdx.x;
  const float* P   = args.P[d];
  const float* Whh = args.Whh[d];
  float* Hout = args.Hout[d];
  const int back = args.back[d];

  const int tid  = threadIdx.x;
  const int j    = tid & (HDIM - 1);
  const int half = tid >> 9;                                  // 0: i/gg, 1: f/o
  const int row0 = half ? (HDIM + j) : j;                     // f_j | i_j
  const int row1 = half ? (3 * HDIM + j) : (2 * HDIM + j);    // o_j | gg_j
  const float4* w04 = (const float4*)(Whh + (long)row0 * HDIM);
  const float4* w14 = (const float4*)(Whh + (long)row1 * HDIM);
  const float4* h4  = (const float4*)hbuf;

  if (tid < HDIM) hbuf[tid] = 0.f;
  float c = 0.f;
  __syncthreads();

  for (int step = 0; step < LEN; ++step) {
    const int t = back ? (LEN - 1 - step) : step;
    const float* pr = P + (long)t * GDIM;
    const int tn = back ? (t - 1) : (t + 1);
    if (tn >= 0 && tn < LEN) __builtin_prefetch(&P[(long)tn * GDIM + row0], 0, 1);

    float d0 = pr[row0], d1 = pr[row1];
#pragma unroll 2
    for (int h = 0; h < HDIM / 4; ++h) {
      float4 hv = h4[h];
      float4 a = w04[h];
      float4 b = w14[h];
      d0 = fmaf(a.x, hv.x, fmaf(a.y, hv.y, fmaf(a.z, hv.z, fmaf(a.w, hv.w, d0))));
      d1 = fmaf(b.x, hv.x, fmaf(b.y, hv.y, fmaf(b.z, hv.z, fmaf(b.w, hv.w, d1))));
    }
    if (half == 0) { gI[j] = d0; gG[j] = d1; }
    else           { gF[j] = d0; gO[j] = d1; }
    __syncthreads();
    if (half == 0) {
      float iv = sigm(gI[j]);
      float fv = sigm(gF[j]);
      float gv = tanhf(gG[j]);
      float ov = sigm(gO[j]);
      c = fv * c + iv * gv;
      float hv = ov * tanhf(c);
      hbuf[j] = hv;
      Hout[(long)t * HDIM + j] = hv;
    }
    __syncthreads();
  }
}

// ---------------------------------------------------------------------------
// 6) Row norms (clamped) for q1p,q2p,q1n,q2n
// ---------------------------------------------------------------------------
struct RNArgs { const float* M[4]; float* N[4]; };

__global__ void rownorm_kernel(RNArgs args) {
  int idx = blockIdx.x * blockDim.x + threadIdx.x;
  if (idx >= 4 * LEN) return;
  int m = idx / LEN, i = idx - m * LEN;
  const float4* row = (const float4*)(args.M[m] + (long)i * HDIM);
  float s = 0.f;
  for (int h = 0; h < HDIM / 4; ++h) {
    float4 v = row[h];
    s = fmaf(v.x, v.x, fmaf(v.y, v.y, fmaf(v.z, v.z, fmaf(v.w, v.w, s))));
  }
  args.N[m][i] = fmaxf(sqrtf(s), EPSF);
}

// 7) Weighted norms n[i,k] = max(||M[i]*W[k]||, EPS)
struct WNArgs { const float* M[4]; const float* W[4]; float* N[4]; };

__global__ void wnorm_kernel(WNArgs args) {
  int idx = blockIdx.x * blockDim.x + threadIdx.x;
  if (idx >= 4 * LEN * WROWS) return;
  int c = idx / (LEN * WROWS);
  int rem = idx - c * (LEN * WROWS);
  int i = rem / WROWS, k = rem - i * WROWS;
  const float4* row = (const float4*)(args.M[c] + (long)i * HDIM);
  const float4* w   = (const float4*)(args.W[c] + (long)k * HDIM);
  float s = 0.f;
  for (int h = 0; h < HDIM / 4; ++h) {
    float4 a = row[h], b = w[h];
    float t0 = a.x * b.x, t1 = a.y * b.y, t2 = a.z * b.z, t3 = a.w * b.w;
    s = fmaf(t0, t0, fmaf(t1, t1, fmaf(t2, t2, fmaf(t3, t3, s))));
  }
  args.N[c][i * WROWS + k] = fmaxf(sqrtf(s), EPSF);
}

// ---------------------------------------------------------------------------
// 8) cos_pair via WMMA: block = one i-tile, sweeps all j-tiles with hoisted A.
// ---------------------------------------------------------------------------
struct CosArgs { const _Float16* Ah[2]; const _Float16* Bh[2];
                 const float* nA[2]; const float* nB[2]; float* out[2]; };

__global__ __launch_bounds__(32) void cospair_kernel(CosArgs args) {
  const int lane = threadIdx.x;
  const int it = blockIdx.x, z = blockIdx.y;
  const _Float16* arow = args.Ah[z] + (it * 16 + (lane & 15)) * HDIM;
  v16h af[16];
#pragma unroll
  for (int kk = 0; kk < 16; ++kk) af[kk] = frag_a_h(arow, kk * 32, lane);
  const int M0 = (lane & 16) ? 8 : 0;
  float inA[8];
#pragma unroll
  for (int v = 0; v < 8; ++v) inA[v] = 1.0f / args.nA[z][it * 16 + M0 + v];
  float* out = args.out[z];
  for (int jt = 0; jt < LEN / 16; ++jt) {
    const _Float16* brow = args.Bh[z] + (jt * 16 + (lane & 15)) * HDIM;
    v8f acc = {};
#pragma unroll
    for (int kk = 0; kk < 16; ++kk)
      acc = wmma_step(af[kk], frag_b_h(brow, kk * 32, lane), acc);
    const int jj = jt * 16 + (lane & 15);
    const float inb = 1.0f / args.nB[z][jj];
#pragma unroll
    for (int v = 0; v < 8; ++v)
      out[(long)(it * 16 + M0 + v) * LEN + jj] = acc[v] * inb * inA[v];
  }
}

// ---------------------------------------------------------------------------
// 9) Attention: att[i,h] = (sum_j alpha[i,j]*q2[j,h]) / (sum_j alpha[i,j])
// ---------------------------------------------------------------------------
struct AttArgs { const float* alpha[2]; const float* Q2[2]; float* att[2]; };

__global__ void att_kernel(AttArgs args) {
  int idx = blockIdx.x * blockDim.x + threadIdx.x;
  if (idx >= 2 * LEN * HDIM) return;
  int z = idx / (LEN * HDIM);
  int rem = idx - z * (LEN * HDIM);
  int i = rem >> 9, h = rem & (HDIM - 1);
  const float* al = args.alpha[z] + (long)i * LEN;
  const float* q  = args.Q2[z];
  float s = 0.f, r = 0.f;
  for (int jj = 0; jj < LEN; ++jj) {
    float a = al[jj];
    s = fmaf(a, q[(long)jj * HDIM + h], s);
    r += a;
  }
  args.att[z][(long)i * HDIM + h] = s / r;
}

// ---------------------------------------------------------------------------
// 10) pair-max: num = (q1*w_k^2) x q2^T via WMMA (w^2 folded into A side),
//     max over j fused in epilogue. Hoisted scaled A fragments.
// ---------------------------------------------------------------------------
struct PMArgs { const float* A[2]; const _Float16* Bh[2]; const float* W[2];
                const float* nA[2]; const float* nB[2]; float* out[2]; };

__global__ __launch_bounds__(32) void pairmax_kernel(PMArgs args) {
  __shared__ float red[32][8];
  const int lane = threadIdx.x;
  const int it = blockIdx.x, k = blockIdx.y, z = blockIdx.z;
  const float* arow = args.A[z] + (long)(it * 16 + (lane & 15)) * HDIM;
  const float* w    = args.W[z] + (long)k * HDIM;
  const float* nB   = args.nB[z];

  v16h af[16];
#pragma unroll
  for (int kk = 0; kk < 16; ++kk) af[kk] = frag_a_w2(arow, w, kk * 32, lane);

  float m[8];
#pragma unroll
  for (int v = 0; v < 8; ++v) m[v] = -3.4e38f;

  for (int jt = 0; jt < LEN / 16; ++jt) {
    const _Float16* brow = args.Bh[z] + (jt * 16 + (lane & 15)) * HDIM;
    v8f acc = {};
#pragma unroll
    for (int kk = 0; kk < 16; ++kk)
      acc = wmma_step(af[kk], frag_b_h(brow, kk * 32, lane), acc);
    const int jj = jt * 16 + (lane & 15);
    const float inb = 1.0f / nB[jj * WROWS + k];
#pragma unroll
    for (int v = 0; v < 8; ++v) m[v] = fmaxf(m[v], acc[v] * inb);
  }
#pragma unroll
  for (int v = 0; v < 8; ++v) red[lane][v] = m[v];
  __syncthreads();
  if (lane < 16) {
    const int M  = lane;
    const int v  = (M < 8) ? M : (M - 8);
    const int lo = (M < 8) ? 0 : 16;
    float mx = -3.4e38f;
#pragma unroll
    for (int l = 0; l < 16; ++l) mx = fmaxf(mx, red[lo + l][v]);
    const int i = it * 16 + M;
    args.out[z][i * WROWS + k] = mx / args.nA[z][i * WROWS + k];
  }
}

// ---------------------------------------------------------------------------
// 11) wcos_rows -> slots {0: pos_full, 1: neg_full, 4: pos_atten, 5: neg_atten}
// ---------------------------------------------------------------------------
struct WRArgs { const float* A[4]; const float* B[4]; const float* W[4];
                int perI[4]; int slot[4]; float* out; };

__global__ void wcos_rows_kernel(WRArgs args) {
  int idx = blockIdx.x * blockDim.x + threadIdx.x;
  if (idx >= 4 * LEN * WROWS) return;
  int m = idx / (LEN * WROWS);
  int rem = idx - m * (LEN * WROWS);
  int i = rem / WROWS, k = rem - i * WROWS;
  const float4* a = (const float4*)(args.A[m] + (long)i * HDIM);
  const float4* b = (const float4*)(args.B[m] + (args.perI[m] ? (long)i * HDIM : 0));
  const float4* w = (const float4*)(args.W[m] + (long)k * HDIM);
  float num = 0.f, na = 0.f, nb = 0.f;
  for (int h = 0; h < HDIM / 4; ++h) {
    float4 av = a[h], bv = b[h], wv = w[h];
    float a0 = av.x * wv.x, a1 = av.y * wv.y, a2 = av.z * wv.z, a3 = av.w * wv.w;
    float b0 = bv.x * wv.x, b1 = bv.y * wv.y, b2 = bv.z * wv.z, b3 = bv.w * wv.w;
    num = fmaf(a0, b0, fmaf(a1, b1, fmaf(a2, b2, fmaf(a3, b3, num))));
    na  = fmaf(a0, a0, fmaf(a1, a1, fmaf(a2, a2, fmaf(a3, a3, na))));
    nb  = fmaf(b0, b0, fmaf(b1, b1, fmaf(b2, b2, fmaf(b3, b3, nb))));
  }
  float den = fmaxf(sqrtf(na), EPSF) * fmaxf(sqrtf(nb), EPSF);
  args.out[(long)args.slot[m] * LEN * WROWS + i * WROWS + k] = num / den;
}

// ---------------------------------------------------------------------------
// Host launcher
// ---------------------------------------------------------------------------
extern "C" void kernel_launch(void* const* d_in, const int* in_sizes, int n_in,
                              void* d_out, int out_size, void* d_ws, size_t ws_size,
                              hipStream_t stream) {
  const int*   q1_ids  = (const int*)d_in[0];
  const int*   q2_ids  = (const int*)d_in[1];
  const float* emb     = (const float*)d_in[4];
  const float* q1Wih_f = (const float*)d_in[5];
  const float* q1Whh_f = (const float*)d_in[6];
  const float* q1Wih_b = (const float*)d_in[7];
  const float* q1Whh_b = (const float*)d_in[8];
  const float* q2Wih_f = (const float*)d_in[9];
  const float* q2Whh_f = (const float*)d_in[10];
  const float* q2Wih_b = (const float*)d_in[11];
  const float* q2Whh_b = (const float*)d_in[12];
  const float* W1 = (const float*)d_in[13];
  const float* W2 = (const float*)d_in[14];
  const float* W3 = (const float*)d_in[15];
  const float* W4 = (const float*)d_in[16];
  const float* W5 = (const float*)d_in[17];
  float* out = (float*)d_out;

  // Workspace carving (all offsets 64B aligned)
  char* cur = (char*)d_ws;
  auto carveF = [&](size_t nfloat) { float* p = (float*)cur; cur += ((nfloat * 4 + 63) & ~63ul); return p; };
  auto carveH = [&](size_t nhalf)  { _Float16* p = (_Float16*)cur; cur += ((nhalf * 2 + 63) & ~63ul); return p; };

  _Float16* xh1 = carveH(LEN * KP);
  _Float16* xh2 = carveH(LEN * KP);
  _Float16* wh[4];
  for (int d = 0; d < 4; ++d) wh[d] = carveH((size_t)GDIM * KP);
  float* P[4];
  for (int d = 0; d < 4; ++d) P[d] = carveF((size_t)LEN * GDIM);
  float* q1p = carveF(LEN * HDIM);
  float* q1n = carveF(LEN * HDIM);
  float* q2p = carveF(LEN * HDIM);
  float* q2n = carveF(LEN * HDIM);
  _Float16* q1ph = carveH(LEN * HDIM);
  _Float16* q1nh = carveH(LEN * HDIM);
  _Float16* q2ph = carveH(LEN * HDIM);
  _Float16* q2nh = carveH(LEN * HDIM);
  float* alp = carveF(LEN * LEN);
  float* aln = carveF(LEN * LEN);
  float* atp = carveF(LEN * HDIM);
  float* atn = carveF(LEN * HDIM);
  float* rn  = carveF(4 * LEN);
  float* wn  = carveF(4 * LEN * WROWS);

  // 1) gather + pack X
  gather_pack_kernel<<<(2 * LEN * KP + 255) / 256, 256, 0, stream>>>(
      emb, q1_ids, q2_ids, xh1, xh2);
  // 2) pack Wih
  {
    PackWArgs a;
    a.src[0] = q1Wih_f; a.src[1] = q1Wih_b; a.src[2] = q2Wih_f; a.src[3] = q2Wih_b;
    for (int d = 0; d < 4; ++d) a.dst[d] = wh[d];
    pack_wih_kernel<<<(4 * GDIM * KP + 255) / 256, 256, 0, stream>>>(a);
  }
  // 3) input projections (WMMA)
  {
    ProjArgs a;
    a.X[0] = xh1; a.X[1] = xh1; a.X[2] = xh2; a.X[3] = xh2;
    for (int d = 0; d < 4; ++d) { a.Wt[d] = wh[d]; a.P[d] = P[d]; }
    proj_gemm_kernel<<<dim3(GDIM / 16, LEN / 16, 4), 32, 0, stream>>>(a);
  }
  // 4) LSTM scans
  {
    LstmArgs a;
    for (int d = 0; d < 4; ++d) a.P[d] = P[d];
    a.Whh[0] = q1Whh_f; a.Whh[1] = q1Whh_b; a.Whh[2] = q2Whh_f; a.Whh[3] = q2Whh_b;
    a.Hout[0] = q1p; a.Hout[1] = q1n; a.Hout[2] = q2p; a.Hout[3] = q2n;
    a.back[0] = 0; a.back[1] = 1; a.back[2] = 0; a.back[3] = 1;
    lstm_kernel<<<4, 1024, 0, stream>>>(a);
  }
  // 5) pack q outputs to f16
  {
    PackQArgs a;
    a.src[0] = q1p; a.src[1] = q1n; a.src[2] = q2p; a.src[3] = q2n;
    a.dst[0] = q1ph; a.dst[1] = q1nh; a.dst[2] = q2ph; a.dst[3] = q2nh;
    pack_q_kernel<<<(4 * LEN * HDIM + 255) / 256, 256, 0, stream>>>(a);
  }
  // 6) row norms
  {
    RNArgs a;
    a.M[0] = q1p; a.M[1] = q2p; a.M[2] = q1n; a.M[3] = q2n;
    a.N[0] = rn; a.N[1] = rn + LEN; a.N[2] = rn + 2 * LEN; a.N[3] = rn + 3 * LEN;
    rownorm_kernel<<<(4 * LEN + 255) / 256, 256, 0, stream>>>(a);
  }
  // 7) weighted norms
  {
    WNArgs a;
    a.M[0] = q1p; a.W[0] = W3; a.N[0] = wn;                   // nA pos
    a.M[1] = q2p; a.W[1] = W3; a.N[1] = wn + LEN * WROWS;     // nB pos
    a.M[2] = q1n; a.W[2] = W4; a.N[2] = wn + 2 * LEN * WROWS; // nA neg
    a.M[3] = q2n; a.W[3] = W4; a.N[3] = wn + 3 * LEN * WROWS; // nB neg
    wnorm_kernel<<<(4 * LEN * WROWS + 255) / 256, 256, 0, stream>>>(a);
  }
  // 8) cos_pair (WMMA)
  {
    CosArgs a;
    a.Ah[0] = q1ph; a.Bh[0] = q2ph; a.nA[0] = rn;           a.nB[0] = rn + LEN;     a.out[0] = alp;
    a.Ah[1] = q1nh; a.Bh[1] = q2nh; a.nA[1] = rn + 2 * LEN; a.nB[1] = rn + 3 * LEN; a.out[1] = aln;
    cospair_kernel<<<dim3(LEN / 16, 2), 32, 0, stream>>>(a);
  }
  // 9) attention
  {
    AttArgs a;
    a.alpha[0] = alp; a.Q2[0] = q2p; a.att[0] = atp;
    a.alpha[1] = aln; a.Q2[1] = q2n; a.att[1] = atn;
    att_kernel<<<(2 * LEN * HDIM + 255) / 256, 256, 0, stream>>>(a);
  }
  // 10) pair-max (WMMA + fused max) -> slots 2, 3
  {
    PMArgs a;
    a.A[0] = q1p; a.Bh[0] = q2ph; a.W[0] = W3;
    a.nA[0] = wn; a.nB[0] = wn + LEN * WROWS;
    a.out[0] = out + (long)2 * LEN * WROWS;
    a.A[1] = q1n; a.Bh[1] = q2nh; a.W[1] = W4;
    a.nA[1] = wn + 2 * LEN * WROWS; a.nB[1] = wn + 3 * LEN * WROWS;
    a.out[1] = out + (long)3 * LEN * WROWS;
    pairmax_kernel<<<dim3(LEN / 16, WROWS, 2), 32, 0, stream>>>(a);
  }
  // 11) wcos_rows -> slots 0, 1, 4, 5
  {
    WRArgs a;
    a.A[0] = q1p; a.B[0] = q2p + (long)(LEN - 1) * HDIM; a.W[0] = W1; a.perI[0] = 0; a.slot[0] = 0;
    a.A[1] = q1n; a.B[1] = q2n;                          a.W[1] = W2; a.perI[1] = 0; a.slot[1] = 1;
    a.A[2] = q1p; a.B[2] = atp;                          a.W[2] = W5; a.perI[2] = 1; a.slot[2] = 4;
    a.A[3] = q1n; a.B[3] = atn;                          a.W[3] = W5; a.perI[3] = 1; a.slot[3] = 5;
    a.out = out;
    wcos_rows_kernel<<<(4 * LEN * WROWS + 255) / 256, 256, 0, stream>>>(a);
  }
  (void)in_sizes; (void)n_in; (void)out_size; (void)ws_size;
}
